// AdaptiveTensorUnit_82712480186467
// MI455X (gfx1250) — matrix operand order repaired and checked by reading, not compiled, not run on hardware
//
#include <hip/hip_runtime.h>
#include <hip/hip_bf16.h>

typedef _Float16 v16h __attribute__((ext_vector_type(16)));
typedef _Float16 h8   __attribute__((ext_vector_type(8)));
typedef float    v8f  __attribute__((ext_vector_type(8)));

#define GSZ 128
#define DIM 128
#define WAVES 4

// ---------------- prologue: fp32 weights -> f16, transposed ----------------
// w1t[n][k] = W1[k][n]  (n<128, k<256)   w2t[n][k] = W2[k][n]  (n,k<128)
__global__ void atu_convert_weights(const float* __restrict__ W1,
                                    const float* __restrict__ W2,
                                    _Float16* __restrict__ w1t,
                                    _Float16* __restrict__ w2t) {
    int idx = blockIdx.x * blockDim.x + threadIdx.x;
    if (idx < 2 * DIM * DIM) {                 // 32768 elements of W1t
        int n = idx >> 8;                      // /256
        int k = idx & 255;
        w1t[n * 256 + k] = (_Float16)W1[k * DIM + n];
    } else if (idx < 2 * DIM * DIM + DIM * DIM) {
        int i2 = idx - 2 * DIM * DIM;
        int n = i2 >> 7;
        int k = i2 & 127;
        w2t[n * DIM + k] = (_Float16)W2[k * DIM + n];
    }
}

__device__ __forceinline__ float fast_tanh(float x) {
    float ax = __builtin_fabsf(x);
    float e  = __expf(-2.0f * ax);
    float t  = (1.0f - e) / (1.0f + e);
    return __builtin_copysignf(t, x);
}

__device__ __forceinline__ void cvt4(v16h& af, int base, float4 a) {
    af[base + 0] = (_Float16)a.x;
    af[base + 1] = (_Float16)a.y;
    af[base + 2] = (_Float16)a.z;
    af[base + 3] = (_Float16)a.w;
}

// ---------------- main fused kernel: one wave == one 16-unit tile ----------
__global__ __launch_bounds__(WAVES * 32, 4)
void atu_kernel(const float* __restrict__ field,
                const float* __restrict__ positions,
                const float* __restrict__ signatures,
                const float* __restrict__ offsets,
                const _Float16* __restrict__ w1t,
                const float* __restrict__ b1,
                const _Float16* __restrict__ w2t,
                const float* __restrict__ b2,
                float* __restrict__ out_stab,
                float* __restrict__ out_pos,
                int N) {
    __shared__ float    sigst[WAVES][16][132];   // fp32 signatures (padded rows)
    __shared__ _Float16 hstage[WAVES][16][136];  // f16 hidden restage (padded rows)
    __shared__ float    stabbuf[WAVES][2][16];
    __shared__ int      offtab[DIM];             // packed neighborhood offsets

    const int tid = threadIdx.x;
    // build first-128 offsets of the 7x7x7 cube, i-major order, packed (+3 bias)
    if (tid < DIM) {
        int n   = tid;
        int oi  = n / 49;
        int rem = n - oi * 49;
        int oj  = rem / 7;
        int ok  = rem - oj * 7;
        offtab[n] = oi | (oj << 8) | (ok << 16);
    }
    __syncthreads();

    const int w    = tid >> 5;
    const int lane = tid & 31;
    const int mlo  = lane & 15;   // M index / column lane
    const int hi   = lane >> 4;   // 0 = low K half, 1 = high K half
    const int tile = blockIdx.x * WAVES + w;
    const int u0   = tile * 16;
    if (u0 >= N) return;
    const int unit = u0 + mlo;

    // ---- stage this tile's signatures (fp32) into LDS, coalesced ----
    {
        const float4* sp = (const float4*)(signatures + (size_t)u0 * DIM);
        #pragma unroll
        for (int i = 0; i < 16; ++i) {
            int f    = lane + 32 * i;          // 0..511 float4s
            float4 v = sp[f];
            int row = f >> 5;
            int c4  = f & 31;
            *(float4*)&sigst[w][row][c4 * 4] = v;
        }
    }

    // per-lane column biases (column n = t*16 + mlo)
    float b1v[8], b2v[8];
    #pragma unroll
    for (int t = 0; t < 8; ++t) {
        b1v[t] = b1[t * 16 + mlo];
        b2v[t] = b2[t * 16 + mlo];
    }

    // ---- pass-invariant signature A-fragments, kept in registers (32 VGPRs) ----
    v16h af_sig[4];
    #pragma unroll
    for (int c = 0; c < 4; ++c) {
        const int k0 = c * 32 + hi * 8;
        const float* srow = &sigst[w][mlo][0];
        cvt4(af_sig[c], 0,  *(const float4*)&srow[k0]);
        cvt4(af_sig[c], 4,  *(const float4*)&srow[k0 + 4]);
        cvt4(af_sig[c], 8,  *(const float4*)&srow[k0 + 16]);
        cvt4(af_sig[c], 12, *(const float4*)&srow[k0 + 20]);
    }

    const float px = positions[unit * 3 + 0];
    const float py = positions[unit * 3 + 1];
    const float pz = positions[unit * 3 + 2];
    const float ox = offsets[unit * 3 + 0];
    const float oy = offsets[unit * 3 + 1];
    const float oz = offsets[unit * 3 + 2];

    #pragma clang loop unroll(disable)
    for (int p = 0; p < 2; ++p) {                 // pass 0: old pos, pass 1: test pos
        // fence: stop LICM from hoisting the (pass-invariant) weight-fragment
        // loads out of this loop — that hoisting costs ~768 VGPRs and tanks
        // occupancy; re-fetching them from WGP$/L1 per pass is far cheaper.
        __asm__ volatile("" ::: "memory");

        const float fx = p ? (px + ox) : px;
        const float fy = p ? (py + oy) : py;
        const float fz = p ? (pz + oz) : pz;
        const int pci = min(max((int)fx, 0), GSZ - 1);
        const int pcj = min(max((int)fy, 0), GSZ - 1);
        const int pck = min(max((int)fz, 0), GSZ - 1);

        // ---------------- GEMM1: [16x256] @ W1 -> h_pre [16x128] ----------------
        v8f acc[8] = {};
        #pragma unroll
        for (int c = 0; c < 8; ++c) {
            v16h af;
            if (c < 4) {
                af = af_sig[c];                 // signature half, precomputed
            } else {
                // local-field half: per-lane gathers straight into the fragment
                const int nb0 = (c - 4) * 32 + hi * 8;
                #pragma unroll
                for (int e = 0; e < 8; ++e) {
                    int o  = offtab[nb0 + e];
                    int xx = min(max(pci + (o & 255) - 3, 0), GSZ - 1);
                    int yy = min(max(pcj + ((o >> 8) & 255) - 3, 0), GSZ - 1);
                    int zz = min(max(pck + ((o >> 16) & 255) - 3, 0), GSZ - 1);
                    af[e] = (_Float16)field[(xx << 14) + (yy << 7) + zz];
                    o  = offtab[nb0 + 16 + e];
                    xx = min(max(pci + (o & 255) - 3, 0), GSZ - 1);
                    yy = min(max(pcj + ((o >> 8) & 255) - 3, 0), GSZ - 1);
                    zz = min(max(pck + ((o >> 16) & 255) - 3, 0), GSZ - 1);
                    af[8 + e] = (_Float16)field[(xx << 14) + (yy << 7) + zz];
                }
            }
            #pragma unroll
            for (int t = 0; t < 8; ++t) {
                const _Float16* bp = w1t + (size_t)(t * 16 + mlo) * 256 + c * 32 + hi * 8;
                v16h bf = __builtin_shufflevector(*(const h8*)bp, *(const h8*)(bp + 16), 0, 1, 2, 3, 4, 5, 6, 7, 8, 9, 10, 11, 12, 13, 14, 15);
                acc[t] = __builtin_amdgcn_wmma_f32_16x16x32_f16(
                    false, af, false, bf, (short)0, acc[t], false, false);
            }
        }

        // ---- bias + tanh, restage h (C layout -> row-major f16 in LDS) ----
        #pragma unroll
        for (int t = 0; t < 8; ++t) {
            #pragma unroll
            for (int e = 0; e < 8; ++e) {
                float v = fast_tanh(acc[t][e] + b1v[t]);
                hstage[w][hi * 8 + e][t * 16 + mlo] = (_Float16)v;
            }
        }
        __asm__ volatile("s_wait_dscnt 0" ::: "memory");   // same-wave LDS fence

        // ---------------- GEMM2: h [16x128] @ W2 -> resp [16x128] ----------------
        v8f acc2[8] = {};
        #pragma unroll
        for (int c = 0; c < 4; ++c) {
            const int k0 = c * 32 + hi * 8;
            h8 alo = *(const h8*)&hstage[w][mlo][k0];
            h8 ahi = *(const h8*)&hstage[w][mlo][k0 + 16];
            v16h af;
            #pragma unroll
            for (int e = 0; e < 8; ++e) { af[e] = alo[e]; af[8 + e] = ahi[e]; }
            #pragma unroll
            for (int t = 0; t < 8; ++t) {
                const _Float16* bp = w2t + (size_t)(t * 16 + mlo) * DIM + c * 32 + hi * 8;
                v16h bf = __builtin_shufflevector(*(const h8*)bp, *(const h8*)(bp + 16), 0, 1, 2, 3, 4, 5, 6, 7, 8, 9, 10, 11, 12, 13, 14, 15);
                acc2[t] = __builtin_amdgcn_wmma_f32_16x16x32_f16(
                    false, af, false, bf, (short)0, acc2[t], false, false);
            }
        }

        // ---- ||resp - sig|| per row: per-lane partials + shfl_xor reduce ----
        float rs[8];
        #pragma unroll
        for (int e = 0; e < 8; ++e) rs[e] = 0.0f;
        #pragma unroll
        for (int t = 0; t < 8; ++t) {
            const int n = t * 16 + mlo;
            #pragma unroll
            for (int e = 0; e < 8; ++e) {
                float d = acc2[t][e] + b2v[t] - sigst[w][hi * 8 + e][n];
                rs[e] += d * d;
            }
        }
        #pragma unroll
        for (int e = 0; e < 8; ++e) {
            rs[e] += __shfl_xor(rs[e], 1, 32);
            rs[e] += __shfl_xor(rs[e], 2, 32);
            rs[e] += __shfl_xor(rs[e], 4, 32);
            rs[e] += __shfl_xor(rs[e], 8, 32);
        }
        if (mlo == 0) {
            #pragma unroll
            for (int e = 0; e < 8; ++e)
                stabbuf[w][p][hi * 8 + e] = __builtin_sqrtf(rs[e]);
        }
    }
    __asm__ volatile("s_wait_dscnt 0" ::: "memory");

    // ---- accept/reject + outputs (one unit per low lane) ----
    if (lane < 16 && unit < N) {
        const float so = stabbuf[w][0][mlo];
        const float sn = stabbuf[w][1][mlo];
        const bool acc_m = (sn <= so);
        out_stab[unit] = acc_m ? sn : so;
        out_pos[unit * 3 + 0] = acc_m ? (px + ox) : px;
        out_pos[unit * 3 + 1] = acc_m ? (py + oy) : py;
        out_pos[unit * 3 + 2] = acc_m ? (pz + oz) : pz;
    }
}

extern "C" void kernel_launch(void* const* d_in, const int* in_sizes, int n_in,
                              void* d_out, int out_size, void* d_ws, size_t ws_size,
                              hipStream_t stream) {
    const float* field      = (const float*)d_in[0];   // [128,128,128]
    const float* positions  = (const float*)d_in[1];   // [N,3]
    const float* signatures = (const float*)d_in[2];   // [N,128]
    const float* offsets    = (const float*)d_in[3];   // [N,3]
    const float* W1         = (const float*)d_in[4];   // [256,128]
    const float* b1         = (const float*)d_in[5];   // [128]
    const float* W2         = (const float*)d_in[6];   // [128,128]
    const float* b2         = (const float*)d_in[7];   // [128]

    const int N = in_sizes[1] / 3;

    // workspace: W1t f16 (64 KB) | W2t f16 (32 KB)
    _Float16* w1t = (_Float16*)d_ws;
    _Float16* w2t = (_Float16*)((char*)d_ws + (size_t)2 * DIM * DIM * sizeof(_Float16));

    const int convN = 2 * DIM * DIM + DIM * DIM;       // 49152
    atu_convert_weights<<<(convN + 255) / 256, 256, 0, stream>>>(W1, W2, w1t, w2t);

    const int tiles  = (N + 15) / 16;
    const int blocks = (tiles + WAVES - 1) / WAVES;
    atu_kernel<<<blocks, WAVES * 32, 0, stream>>>(
        field, positions, signatures, offsets, w1t, b1, w2t, b2,
        (float*)d_out, (float*)d_out + N, N);
}
